// CustomMLP_14791867368206
// MI455X (gfx1250) — matrix operand order repaired and checked by reading, not compiled, not run on hardware
//
#include <hip/hip_runtime.h>
#include <cmath>

// ---------------------------------------------------------------------------
// MLP: out = GELU(x @ W1 + b1) @ W2 + b2
// x:[8192,1024] f32, W1:[1024,4096], W2:[4096,1024], out:[8192,1024] f32
//
// Phase 0: one-shot fp32->bf16 conversion of X (row-major) and W1/W2
//          (transposed to [N][K]) -- memory-bound prelude (~5us at 23.3TB/s).
// Phase 1/2: bf16 WMMA GEMMs whose inner loop is pure
//          global_load_async_to_lds_b128 (ASYNCcnt) + ds_load_b128 + v_wmma,
//          double-buffered so the async DMA of tile t+1 overlaps WMMA of t.
// ---------------------------------------------------------------------------

typedef __attribute__((ext_vector_type(16))) __bf16 v16bf;
typedef __attribute__((ext_vector_type(8)))  float  v8f;
typedef __attribute__((ext_vector_type(2)))  __bf16 v2bf;
typedef __attribute__((ext_vector_type(2)))  float  v2f;

#define BM 128
#define BN 128
#define BK 32
#define LDK 40   // padded LDS K-stride in ushorts (80 B, keeps 16B alignment)

static __device__ __forceinline__ unsigned short bf16_bits(float f) {
    union { __bf16 h; unsigned short s; } u;
    u.h = (__bf16)f;
    return u.s;
}
static __device__ __forceinline__ unsigned int pack2_bf16(float a, float b) {
    v2f f = {a, b};
    union { v2bf h; unsigned int u; } c;
    c.h = __builtin_convertvector(f, v2bf);   // -> v_cvt_pk_bf16_f32
    return c.u;
}

union Frag {
    v16bf v;
    uint4 u[2];
};

// ---- gfx1250 async global->LDS copy (16 B per lane), tracked by ASYNCcnt ---
static __device__ __forceinline__ void async_copy_b128(unsigned lds_off,
                                                       const void* gptr) {
    asm volatile("global_load_async_to_lds_b128 %0, %1, off"
                 :: "v"(lds_off), "v"(gptr)
                 : "memory");
}
static __device__ __forceinline__ void wait_asynccnt0() {
    asm volatile("s_wait_asynccnt 0x0" ::: "memory");
}

// ---------------------------------------------------------------------------
// Prelude A: elementwise fp32 -> bf16 (X -> Xb), 4 elems/thread
// ---------------------------------------------------------------------------
__global__ void __launch_bounds__(256)
convert_bf16(const float4* __restrict__ src, uint2* __restrict__ dst, int n4) {
    int i = blockIdx.x * 256 + threadIdx.x;
    if (i < n4) {
        float4 f = src[i];
        uint2 o;
        o.x = pack2_bf16(f.x, f.y);
        o.y = pack2_bf16(f.z, f.w);
        dst[i] = o;
    }
}

// ---------------------------------------------------------------------------
// Prelude B: W[K][N] fp32 -> Wt[N][K] bf16 via 32x32 LDS tile transpose
// grid = (N/32, K/32), block = 256
// ---------------------------------------------------------------------------
__global__ void __launch_bounds__(256)
convert_transpose(const float* __restrict__ W, unsigned short* __restrict__ Wt,
                  int K, int N) {
    __shared__ unsigned short t[32][34];
    const int nb = blockIdx.x * 32;
    const int kb = blockIdx.y * 32;
    const int tx = threadIdx.x & 31;
    const int ty = threadIdx.x >> 5;          // 0..7
#pragma unroll
    for (int r = 0; r < 32; r += 8)
        t[ty + r][tx] = bf16_bits(W[(size_t)(kb + ty + r) * N + nb + tx]);
    __syncthreads();
#pragma unroll
    for (int r = 0; r < 32; r += 8)
        Wt[(size_t)(nb + ty + r) * K + kb + tx] = t[tx][ty + r];
}

// ---------------------------------------------------------------------------
// bf16 WMMA GEMM:  out[M,N] = epilogue( A[M,K] @ Bt[N,K]^T + bias[N] )
//   EPI = 0 : exact-erf GELU, store bf16 (intermediate H)
//   EPI = 1 : store fp32 (final output)
// Block 128x128x32, 8 waves (2x4), wave tile 64x32 = 4x2 WMMA tiles.
// Both LDS tiles staged by async global->LDS DMA, double-buffered.
// ---------------------------------------------------------------------------
template <int EPI>
__global__ void __launch_bounds__(256)
gemm_bf16_wmma(const unsigned short* __restrict__ A,   // [M][K] bf16
               const unsigned short* __restrict__ Bt,  // [N][K] bf16
               const float* __restrict__ bias,
               void* __restrict__ outp,
               int M, int K, int N) {
    __shared__ __align__(16) unsigned short As[2][BM][LDK];
    __shared__ __align__(16) unsigned short Bs[2][BN][LDK];

    const int tid  = threadIdx.x;
    const int lane = tid & 31;
    const int wave = tid >> 5;
    const int wm   = wave >> 2;          // 0..1 -> 64 rows each
    const int wn   = wave & 3;           // 0..3 -> 32 cols each
    const int lm   = lane & 15;
    const int lh   = lane >> 4;

    const int n0 = blockIdx.x * BN;
    const int m0 = blockIdx.y * BM;

    // staging map: 2 threads per row, 16 ushorts (2 x b128) per thread
    const int srow = tid >> 1;            // 0..127
    const int scol = (tid & 1) * 16;      // 0 or 16

    const unsigned short* aptr = A  + (size_t)(m0 + srow) * K + scol;
    const unsigned short* bptr = Bt + (size_t)(n0 + srow) * K + scol;

    const unsigned ldsA0 = (unsigned)(size_t)&As[0][srow][scol];
    const unsigned ldsB0 = (unsigned)(size_t)&Bs[0][srow][scol];
    const unsigned bufA  = (unsigned)sizeof(As[0]);
    const unsigned bufB  = (unsigned)sizeof(Bs[0]);

    v8f acc[4][2];
#pragma unroll
    for (int i = 0; i < 4; ++i)
#pragma unroll
        for (int j = 0; j < 2; ++j) acc[i][j] = 0.0f;

    auto issue = [&](int buf, int k0) {
        const unsigned la = ldsA0 + (unsigned)buf * bufA;
        const unsigned lb = ldsB0 + (unsigned)buf * bufB;
        async_copy_b128(la,      aptr + k0);
        async_copy_b128(la + 16, aptr + k0 + 8);
        async_copy_b128(lb,      bptr + k0);
        async_copy_b128(lb + 16, bptr + k0 + 8);
    };

    issue(0, 0);

    int t = 0;
    for (int k0 = 0; k0 < K; k0 += BK, ++t) {
        const int cur = t & 1;
        wait_asynccnt0();        // this lane's DMA for tile t has landed in LDS
        __syncthreads();         // all lanes' tile-t DMA visible; buf 1-cur free
        if (k0 + BK < K) issue(1 - cur, k0 + BK);   // overlap DMA with WMMA

        Frag af[4], bfr[2];
#pragma unroll
        for (int i = 0; i < 4; ++i) {
            const unsigned short* base = &As[cur][wm * 64 + i * 16 + lm][lh * 8];
            af[i].u[0] = *(const uint4*)base;
            af[i].u[1] = *(const uint4*)(base + 16);
        }
#pragma unroll
        for (int j = 0; j < 2; ++j) {
            const unsigned short* base = &Bs[cur][wn * 32 + j * 16 + lm][lh * 8];
            bfr[j].u[0] = *(const uint4*)base;
            bfr[j].u[1] = *(const uint4*)(base + 16);
        }
#pragma unroll
        for (int i = 0; i < 4; ++i)
#pragma unroll
            for (int j = 0; j < 2; ++j)
                acc[i][j] = __builtin_amdgcn_wmma_f32_16x16x32_bf16(
                    false, af[i].v, false, bfr[j].v, (short)0, acc[i][j],
                    false, false);
    }

    // ---- epilogue --------------------------------------------------------
#pragma unroll
    for (int i = 0; i < 4; ++i) {
#pragma unroll
        for (int j = 0; j < 2; ++j) {
            const int mb = m0 + wm * 64 + i * 16 + lh * 8;
            const int nb = n0 + wn * 32 + j * 16 + lm;
            const float bv = bias[nb];
            if (EPI == 0) {
                unsigned short* H = (unsigned short*)outp;
#pragma unroll
                for (int r = 0; r < 8; ++r) {
                    float v = acc[i][j][r] + bv;
                    float g = 0.5f * v * (1.0f + erff(v * 0.70710678118654752f));
                    H[(size_t)(mb + r) * N + nb] = bf16_bits(g);
                }
            } else {
                float* O = (float*)outp;
#pragma unroll
                for (int r = 0; r < 8; ++r)
                    O[(size_t)(mb + r) * N + nb] = acc[i][j][r] + bv;
            }
        }
    }
}

// ---------------------------------------------------------------------------
extern "C" void kernel_launch(void* const* d_in, const int* in_sizes, int n_in,
                              void* d_out, int out_size, void* d_ws, size_t ws_size,
                              hipStream_t stream) {
    const int M = 8192;      // 4 * 2048 tokens
    const int E = 1024;      // n_embd
    const int S = 4096;      // n_state

    const float* X  = (const float*)d_in[0];
    const float* W1 = (const float*)d_in[1];
    const float* B1 = (const float*)d_in[2];
    const float* W2 = (const float*)d_in[3];
    const float* B2 = (const float*)d_in[4];
    float* OUT = (float*)d_out;

    // workspace layout (bf16 ushorts): H | Xb | W1t | W2t  (~96 MB total)
    unsigned short* Hws = (unsigned short*)d_ws;              // [M][S]
    unsigned short* Xb  = Hws + (size_t)M * S;                // [M][E]
    unsigned short* W1t = Xb  + (size_t)M * E;                // [S][E]
    unsigned short* W2t = W1t + (size_t)S * E;                // [E][S]

    dim3 block(256, 1, 1);

    // Phase 0: conversions
    convert_bf16<<<dim3((M * E / 4 + 255) / 256), block, 0, stream>>>(
        (const float4*)X, (uint2*)Xb, M * E / 4);
    convert_transpose<<<dim3(S / 32, E / 32), block, 0, stream>>>(W1, W1t, E, S);
    convert_transpose<<<dim3(E / 32, S / 32), block, 0, stream>>>(W2, W2t, S, E);

    // Phase 1: H = bf16(GELU(Xb @ W1 + b1))
    gemm_bf16_wmma<0><<<dim3(S / BN, M / BM), block, 0, stream>>>(
        Xb, W1t, B1, (void*)Hws, M, E, S);

    // Phase 2: OUT = H @ W2 + b2
    gemm_bf16_wmma<1><<<dim3(E / BN, M / BM), block, 0, stream>>>(
        Hws, W2t, B2, (void*)OUT, M, S, E);
}
